// AttentionOnlyInteraction_35785667510460
// MI455X (gfx1250) — compile-verified, matrix operand order checked by reading, not run on hardware
//
#include <hip/hip_runtime.h>

// ---------------------------------------------------------------------------
// Problem constants (from reference): B=4, K=1024, D=1024, H=16, dh=64
// ---------------------------------------------------------------------------
#define BATCH 4
#define SEQ   1024
#define DMODEL 1024
#define NHEAD 16
#define DHEAD 64
#define NTOK  (BATCH * SEQ)          // 4096 token rows
#define INV_SCALE 0.125f             // 1/sqrt(64)
#define NEGBIG (-1e9f)

typedef __attribute__((ext_vector_type(16))) __bf16 v16bf;
typedef __attribute__((ext_vector_type(8)))  float  v8f;
typedef unsigned short ushort_t;

// f32 -> bf16 round-to-nearest-even
__device__ __forceinline__ ushort_t f2bf(float x) {
    union { float f; unsigned u; } a; a.f = x;
    unsigned u = a.u;
    unsigned r = u + 0x7fffu + ((u >> 16) & 1u);
    return (ushort_t)(r >> 16);
}

// ---------------------------------------------------------------------------
// WMMA 16x16x32 bf16 fragment loaders.
// 16-bit A-matrix layout (ISA 7.12.2): lane L<16 holds row M=L with
// K = {k0..k0+7, k0+16..k0+23}; lane L>=16 holds row M=L-16 with
// K = {k0+8..k0+15, k0+24..k0+31}. B mirrors this with N as the row index of
// the transposed operand, so both sides use the same loader against
// row-major [row][Kdim] storage.
// ---------------------------------------------------------------------------
__device__ __forceinline__ v16bf gfrag(const ushort_t* __restrict__ p, int ld, int lane) {
    int r  = lane & 15;
    int hi = (lane >> 4) & 1;
    const ushort_t* rp = p + r * ld + hi * 8;
    union { v16bf v; uint4 u[2]; } f;
    f.u[0] = *reinterpret_cast<const uint4*>(rp);        // K 0..7
    f.u[1] = *reinterpret_cast<const uint4*>(rp + 16);   // K 16..23
    return f.v;
}

// prefetch the cachelines a future gfrag at `p` will touch (global_prefetch_b8)
__device__ __forceinline__ void pfrag(const ushort_t* p, int ld, int lane) {
    int r  = lane & 15;
    int hi = (lane >> 4) & 1;
    __builtin_prefetch(p + r * ld + hi * 8, 0, 1);
}

// Same fragment pattern sourced from f32 LDS (softmax probabilities).
__device__ __forceinline__ v16bf lfrag(const float* __restrict__ S, int ld, int k0, int lane) {
    int r  = lane & 15;
    int hi = (lane >> 4) & 1;
    const float* p = S + r * ld + k0 + hi * 8;
    union { v16bf v; ushort_t s[16]; } f;
#pragma unroll
    for (int i = 0; i < 8; ++i) f.s[i]     = f2bf(p[i]);
#pragma unroll
    for (int i = 0; i < 8; ++i) f.s[8 + i] = f2bf(p[16 + i]);
    return f.v;
}

__device__ __forceinline__ v8f wmma_bf16(v16bf a, v16bf b, v8f c) {
    return __builtin_amdgcn_wmma_f32_16x16x32_bf16(false, a, false, b, (short)0, c, false, false);
}

// ---------------------------------------------------------------------------
// Kernel 1: f32 -> bf16 bulk convert (tokens)
// ---------------------------------------------------------------------------
__global__ void k_cvt_bf16(const float* __restrict__ in, ushort_t* __restrict__ out, int n4) {
    int i = (blockIdx.x * blockDim.x + threadIdx.x);
    if (i < n4) {
        float4 v = reinterpret_cast<const float4*>(in)[i];
        ushort_t o0 = f2bf(v.x), o1 = f2bf(v.y), o2 = f2bf(v.z), o3 = f2bf(v.w);
        reinterpret_cast<ushort4*>(out)[i] = make_ushort4(o0, o1, o2, o3);
    }
}

// Kernel 2: transpose + convert: in f32 [Kd][N] -> out bf16 [N][Kd]
__global__ void k_transpose_bf16(const float* __restrict__ in, ushort_t* __restrict__ out,
                                 int Kd, int N) {
    int idx = blockIdx.x * blockDim.x + threadIdx.x;
    if (idx < Kd * N) {
        int k = idx % Kd;
        int n = idx / Kd;
        out[idx] = f2bf(in[k * N + n]);
    }
}

// ---------------------------------------------------------------------------
// Kernel 3: QKV projection GEMM, 32x64 tile per wave, register-double-buffered
// k-loop. C[4096][3072] = tokens_bf @ WqkvT^T + bias, scattered into
//   Q [b*16+h][1024][64] (pre-scaled by 1/sqrt(dh))
//   K [b*16+h][1024][64]
//   Vt[b*16+h][64][1024]  (transposed so PV B-fragments load contiguously)
// ---------------------------------------------------------------------------
__global__ void k_qkv_gemm(const ushort_t* __restrict__ A,      // tokens bf16 [4096][1024]
                           const ushort_t* __restrict__ Bt,     // WqkvT bf16 [3072][1024]
                           const float* __restrict__ bias,      // [3072]
                           ushort_t* __restrict__ Q,
                           ushort_t* __restrict__ Kb,
                           ushort_t* __restrict__ Vt) {
    const int lane = threadIdx.x & 31;
    const int wid  = blockIdx.x * (blockDim.x >> 5) + (threadIdx.x >> 5);
    const int tm   = wid / 48;          // 128 row tiles of 32
    const int tn   = wid % 48;          // 48 col tiles of 64

    v8f acc[2][4] = {};
    const ushort_t* Ap0 = A + (tm * 32 + 0) * DMODEL;
    const ushort_t* Ap1 = A + (tm * 32 + 16) * DMODEL;
    const ushort_t* Bp  = Bt + (tn * 64) * DMODEL;

    // prologue: fragments for k0 = 0
    v16bf a0 = gfrag(Ap0, DMODEL, lane);
    v16bf a1 = gfrag(Ap1, DMODEL, lane);
    v16bf bb[4];
#pragma unroll
    for (int j = 0; j < 4; ++j) bb[j] = gfrag(Bp + j * 16 * DMODEL, DMODEL, lane);

    for (int k0 = 0; k0 < DMODEL; k0 += 32) {
        // issue next-step loads before consuming current fragments
        int kn = (k0 + 32 < DMODEL) ? (k0 + 32) : k0;
        v16bf na0 = gfrag(Ap0 + kn, DMODEL, lane);
        v16bf na1 = gfrag(Ap1 + kn, DMODEL, lane);
        v16bf nb[4];
#pragma unroll
        for (int j = 0; j < 4; ++j) nb[j] = gfrag(Bp + j * 16 * DMODEL + kn, DMODEL, lane);
#pragma unroll
        for (int j = 0; j < 4; ++j) {
            acc[0][j] = wmma_bf16(a0, bb[j], acc[0][j]);
            acc[1][j] = wmma_bf16(a1, bb[j], acc[1][j]);
        }
        a0 = na0; a1 = na1;
#pragma unroll
        for (int j = 0; j < 4; ++j) bb[j] = nb[j];
    }

    const int hi = (lane >> 4) & 1;
    const int nl = lane & 15;
#pragma unroll
    for (int half = 0; half < 2; ++half) {
#pragma unroll
        for (int j = 0; j < 4; ++j) {
#pragma unroll
            for (int r = 0; r < 8; ++r) {
                int m = tm * 32 + half * 16 + r + hi * 8;  // token row
                int n = tn * 64 + j * 16 + nl;             // col in [0,3072)
                float val = acc[half][j][r] + bias[n];
                int which = n >> 10;                       // 0:Q 1:K 2:V
                int d  = n & 1023;
                int h  = d >> 6;
                int di = d & 63;
                int bb2 = m >> 10;
                int kp  = m & 1023;
                int head = bb2 * NHEAD + h;
                if (which == 0) {
                    Q[(head * SEQ + kp) * DHEAD + di] = f2bf(val * INV_SCALE);
                } else if (which == 1) {
                    Kb[(head * SEQ + kp) * DHEAD + di] = f2bf(val);
                } else {
                    Vt[(head * DHEAD + di) * SEQ + kp] = f2bf(val);
                }
            }
        }
    }
}

// ---------------------------------------------------------------------------
// Kernel 4: attention core. One block (8 waves, 256 threads) owns one
// (batch, 16-query tile) and loops over all 16 heads:
//   S = Q K^T (WMMA, masked) -> softmax max/exp in LDS -> head-mean
//   accumulated in a second LDS buffer -> O = P V (WMMA on raw exp, row
//   normalization folded into the O epilogue). No global atomics.
// ---------------------------------------------------------------------------
__global__ void k_attention(const ushort_t* __restrict__ Q,
                            const ushort_t* __restrict__ Kb,
                            const ushort_t* __restrict__ Vt,
                            const int* __restrict__ mask,     // [B][SEQ]
                            float* __restrict__ attn_out,     // [B][SEQ][SEQ]
                            ushort_t* __restrict__ O) {       // [4096][1024] bf16
    __shared__ float S[16 * SEQ];        // 64 KB exp(score) rows
    __shared__ float ASUM[16 * SEQ];     // 64 KB head-mean accumulator
    __shared__ float Osh[16 * DHEAD];    // 4 KB  cross-wave O accumulator
    __shared__ float red[16 * 16];       // softmax partials
    __shared__ float rowinv[16];         // 1/rowsum

    const int tid  = threadIdx.x;
    const int lane = tid & 31;
    const int wave = tid >> 5;              // 0..7
    const int qt   = blockIdx.x & 63;
    const int b    = blockIdx.x >> 6;

    // zero head-mean accumulator (each thread owns a fixed slice)
#pragma unroll 4
    for (int i = 0; i < 64; ++i) ASUM[tid + i * 256] = 0.f;

    const int hi  = (lane >> 4) & 1;
    const int nl  = lane & 15;
    const int row = tid & 15;
    const int seg = tid >> 4;               // 0..15, 64 keys each
    const int* mb = mask + b * SEQ;
    float* Sr = S + row * SEQ + seg * 64;
    float* Ar = ASUM + row * SEQ + seg * 64;

    // query-row mask bits for this lane's 8 accumulator rows (loop-invariant)
    int tmq[8];
#pragma unroll
    for (int r = 0; r < 8; ++r) tmq[r] = mb[qt * 16 + r + hi * 8];

    for (int h = 0; h < NHEAD; ++h) {
        const int head = b * NHEAD + h;
        const int hn   = b * NHEAD + ((h + 1) & 15);   // wrap: harmless prefetch

        // ---- S = (Q/sqrt(dh)) @ K^T, masked, into LDS ---------------------
        const ushort_t* Qp = Q + (head * SEQ + qt * 16) * DHEAD;
        v16bf qa0 = gfrag(Qp + 0,  DHEAD, lane);
        v16bf qa1 = gfrag(Qp + 32, DHEAD, lane);
        // prefetch next head's Q tile and this wave's first K tile
        pfrag(Q + (hn * SEQ + qt * 16) * DHEAD, DHEAD, lane);
        pfrag(Kb + (hn * SEQ + wave * 128) * DHEAD, DHEAD, lane);

        const ushort_t* Kbase = Kb + (head * SEQ + wave * 128) * DHEAD;
        v16bf kb0 = gfrag(Kbase + 0,  DHEAD, lane);
        v16bf kb1 = gfrag(Kbase + 32, DHEAD, lane);
        for (int kk = 0; kk < 8; ++kk) {
            int knext = (kk < 7) ? (kk + 1) : kk;
            v16bf nk0 = gfrag(Kbase + knext * 16 * DHEAD + 0,  DHEAD, lane);
            v16bf nk1 = gfrag(Kbase + knext * 16 * DHEAD + 32, DHEAD, lane);
            v8f s = {};
            s = wmma_bf16(qa0, kb0, s);
            s = wmma_bf16(qa1, kb1, s);
            kb0 = nk0; kb1 = nk1;
            int col = wave * 128 + kk * 16 + nl;
            int tmk = mb[col];
#pragma unroll
            for (int r = 0; r < 8; ++r) {
                float mv = (tmq[r] & tmk) ? 0.f : NEGBIG;
                S[(r + hi * 8) * SEQ + col] = s[r] + mv;
            }
        }
        __syncthreads();

        // ---- softmax: row max ---------------------------------------------
        float mx = -3.4e38f;
#pragma unroll 4
        for (int j = 0; j < 64; ++j) mx = fmaxf(mx, Sr[j]);
        red[row * 16 + seg] = mx;
        __syncthreads();
        float rmx = red[row * 16 + 0];
#pragma unroll
        for (int j = 1; j < 16; ++j) rmx = fmaxf(rmx, red[row * 16 + j]);
        __syncthreads();

        // ---- exp + partial sum (S keeps raw exp; no normalize write) ------
        float sum = 0.f;
#pragma unroll 4
        for (int j = 0; j < 64; ++j) {
            float e = __expf(Sr[j] - rmx);
            Sr[j] = e;
            sum += e;
        }
        red[row * 16 + seg] = sum;
        __syncthreads();
        float rs = 0.f;
#pragma unroll
        for (int j = 0; j < 16; ++j) rs += red[row * 16 + j];
        float inv = 1.f / rs;
        if (seg == 0) rowinv[row] = inv;

        // head-mean accumulation (thread-private slices, no write-back of P)
#pragma unroll 4
        for (int j = 0; j < 64; ++j) Ar[j] += Sr[j] * (inv * (1.0f / NHEAD));

        // zero cross-wave O accumulator for this head
#pragma unroll
        for (int i = 0; i < 4; ++i) Osh[tid + i * 256] = 0.f;
        __syncthreads();   // rowinv + Osh zero visible; S(exp) already synced

        // ---- O = exp(S) @ V : each wave covers 128 keys -------------------
        v8f oacc[4] = {};
        const ushort_t* Vbase = Vt + head * DHEAD * SEQ;
        v16bf vb[4];
#pragma unroll
        for (int j = 0; j < 4; ++j) vb[j] = gfrag(Vbase + j * 16 * SEQ + wave * 128, SEQ, lane);
        for (int c = 0; c < 4; ++c) {
            int k0 = wave * 128 + c * 32;
            int knx = wave * 128 + ((c < 3) ? (c + 1) : c) * 32;
            v16bf nvb[4];
#pragma unroll
            for (int j = 0; j < 4; ++j) nvb[j] = gfrag(Vbase + j * 16 * SEQ + knx, SEQ, lane);
            v16bf pa = lfrag(S, SEQ, k0, lane);
#pragma unroll
            for (int j = 0; j < 4; ++j) oacc[j] = wmma_bf16(pa, vb[j], oacc[j]);
#pragma unroll
            for (int j = 0; j < 4; ++j) vb[j] = nvb[j];
        }
        // fold 1/rowsum into the epilogue
        float rv[8];
#pragma unroll
        for (int r = 0; r < 8; ++r) rv[r] = rowinv[r + hi * 8];
#pragma unroll
        for (int j = 0; j < 4; ++j) {
#pragma unroll
            for (int r = 0; r < 8; ++r) {
                atomicAdd(&Osh[(r + hi * 8) * DHEAD + j * 16 + nl], oacc[j][r] * rv[r]);
            }
        }
        __syncthreads();

        // write merged-head O tile (this head's 64 columns) as bf16
#pragma unroll
        for (int i = 0; i < 4; ++i) {
            int idx = tid + i * 256;
            int m = idx >> 6;
            int n = idx & 63;
            O[((size_t)(b * SEQ + qt * 16 + m)) * DMODEL + h * DHEAD + n] = f2bf(Osh[idx]);
        }
        __syncthreads();   // protect S/Osh before next head overwrites
    }

    // ---- write head-mean attention once, plain coalesced stores -----------
#pragma unroll 4
    for (int i = 0; i < 64; ++i) {
        int idx = tid + i * 256;           // [0, 16*1024)
        int m   = idx >> 10;
        int col = idx & 1023;
        attn_out[((size_t)(b * SEQ + qt * 16 + m)) * SEQ + col] = ASUM[idx];
    }
}

// ---------------------------------------------------------------------------
// Kernel 5: output projection + bias + residual, 32x64 tile per wave,
// register-double-buffered k-loop.
// tokens_out[4096][1024] = O_bf16 @ WprojT^T + b_proj + tokens
// ---------------------------------------------------------------------------
__global__ void k_proj_gemm(const ushort_t* __restrict__ A,    // O bf16 [4096][1024]
                            const ushort_t* __restrict__ Bt,   // WprojT bf16 [1024][1024]
                            const float* __restrict__ bias,    // [1024]
                            const float* __restrict__ resid,   // tokens f32 [4096][1024]
                            float* __restrict__ out) {         // tokens_out f32
    const int lane = threadIdx.x & 31;
    const int wid  = blockIdx.x * (blockDim.x >> 5) + (threadIdx.x >> 5);
    const int tm   = wid >> 4;          // 128 row tiles of 32
    const int tn   = wid & 15;          // 16 col tiles of 64

    v8f acc[2][4] = {};
    const ushort_t* Ap0 = A + (tm * 32 + 0) * DMODEL;
    const ushort_t* Ap1 = A + (tm * 32 + 16) * DMODEL;
    const ushort_t* Bp  = Bt + (tn * 64) * DMODEL;

    v16bf a0 = gfrag(Ap0, DMODEL, lane);
    v16bf a1 = gfrag(Ap1, DMODEL, lane);
    v16bf bb[4];
#pragma unroll
    for (int j = 0; j < 4; ++j) bb[j] = gfrag(Bp + j * 16 * DMODEL, DMODEL, lane);

    for (int k0 = 0; k0 < DMODEL; k0 += 32) {
        int kn = (k0 + 32 < DMODEL) ? (k0 + 32) : k0;
        v16bf na0 = gfrag(Ap0 + kn, DMODEL, lane);
        v16bf na1 = gfrag(Ap1 + kn, DMODEL, lane);
        v16bf nb[4];
#pragma unroll
        for (int j = 0; j < 4; ++j) nb[j] = gfrag(Bp + j * 16 * DMODEL + kn, DMODEL, lane);
#pragma unroll
        for (int j = 0; j < 4; ++j) {
            acc[0][j] = wmma_bf16(a0, bb[j], acc[0][j]);
            acc[1][j] = wmma_bf16(a1, bb[j], acc[1][j]);
        }
        a0 = na0; a1 = na1;
#pragma unroll
        for (int j = 0; j < 4; ++j) bb[j] = nb[j];
    }

    const int hi = (lane >> 4) & 1;
    const int nl = lane & 15;
#pragma unroll
    for (int half = 0; half < 2; ++half) {
#pragma unroll
        for (int j = 0; j < 4; ++j) {
#pragma unroll
            for (int r = 0; r < 8; ++r) {
                int m = tm * 32 + half * 16 + r + hi * 8;
                int n = tn * 64 + j * 16 + nl;
                size_t idx = (size_t)m * DMODEL + n;
                out[idx] = acc[half][j][r] + bias[n] + resid[idx];
            }
        }
    }
}

// ---------------------------------------------------------------------------
// Host-side launch. Workspace layout (bytes):
//   [0,8Mi)     tokens bf16        [8Mi,14Mi)  WqkvT bf16
//   [14Mi,16Mi) WprojT bf16        [16Mi,24Mi) Q bf16
//   [24Mi,32Mi) K bf16             [32Mi,40Mi) Vt bf16
//   [40Mi,48Mi) O bf16             (requires ws_size >= 48 MiB)
// ---------------------------------------------------------------------------
extern "C" void kernel_launch(void* const* d_in, const int* in_sizes, int n_in,
                              void* d_out, int out_size, void* d_ws, size_t ws_size,
                              hipStream_t stream) {
    const float* tokens  = (const float*)d_in[0];
    const int*   tmask   = (const int*)d_in[1];
    const float* W_qkv   = (const float*)d_in[2];
    const float* b_qkv   = (const float*)d_in[3];
    const float* W_proj  = (const float*)d_in[4];
    const float* b_proj  = (const float*)d_in[5];

    float* tokens_out = (float*)d_out;                         // [4][1024][1024]
    float* attn_out   = (float*)d_out + (size_t)NTOK * DMODEL; // [4][1024][1024]

    char* ws = (char*)d_ws;
    const size_t MiB = 1024 * 1024;
    ushort_t* tok_bf = (ushort_t*)(ws + 0);
    ushort_t* WqkvT  = (ushort_t*)(ws + 8 * MiB);
    ushort_t* WprojT = (ushort_t*)(ws + 14 * MiB);
    ushort_t* Qb     = (ushort_t*)(ws + 16 * MiB);
    ushort_t* Kb     = (ushort_t*)(ws + 24 * MiB);
    ushort_t* Vt     = (ushort_t*)(ws + 32 * MiB);
    ushort_t* Ob     = (ushort_t*)(ws + 40 * MiB);

    // 1) conversions / transposes
    k_cvt_bf16<<<(NTOK * DMODEL / 4 + 255) / 256, 256, 0, stream>>>(tokens, tok_bf, NTOK * DMODEL / 4);
    k_transpose_bf16<<<(DMODEL * 3 * DMODEL + 255) / 256, 256, 0, stream>>>(W_qkv, WqkvT, DMODEL, 3 * DMODEL);
    k_transpose_bf16<<<(DMODEL * DMODEL + 255) / 256, 256, 0, stream>>>(W_proj, WprojT, DMODEL, DMODEL);
    // 2) QKV projection: 128 x 48 wave tiles of 32x64, 8 waves per block
    k_qkv_gemm<<<(128 * 48) / 8, 256, 0, stream>>>(tok_bf, WqkvT, b_qkv, Qb, Kb, Vt);
    // 3) attention: one block (8 waves) per (b, 16-query tile), loops all heads
    k_attention<<<BATCH * (SEQ / 16), 256, 0, stream>>>(Qb, Kb, Vt, tmask, attn_out, Ob);
    // 4) output projection + residual: 128 x 16 wave tiles of 32x64
    k_proj_gemm<<<(128 * 16) / 8, 256, 0, stream>>>(Ob, WprojT, b_proj, tokens, tokens_out);
}
